// GridToSequence_10943576670702
// MI455X (gfx1250) — matrix operand order repaired and checked by reading, not compiled, not run on hardware
//
#include <hip/hip_runtime.h>
#include <math.h>

// ---------------------------------------------------------------------------
// Types for CDNA5 WMMA (wave32): v_wmma_f32_16x16x32_bf16
// ---------------------------------------------------------------------------
typedef __attribute__((ext_vector_type(16))) __bf16 v16bf;
typedef __attribute__((ext_vector_type(8)))  float  v8f;
typedef unsigned short u16;

union FragU { v16bf v; uint4 q[2]; };
union AccU  { v8f v; float f[8]; };

__device__ __forceinline__ u16 f32_to_bf16(float f) {
    unsigned int u = __float_as_uint(f);
    unsigned int r = u + 0x7FFFu + ((u >> 16) & 1u);   // round-to-nearest-even
    return (u16)(r >> 16);
}

// A fragment (16x32 bf16, row-major source, ld in elements).
// Lane layout: lanes 0-15 row M=lane, K {k0..k0+7, k0+16..k0+23};
//              lanes 16-31 same rows, K {+8.., +24..}.
__device__ __forceinline__ v16bf load_frag_a(const u16* base, int ld, int row, int k0, int lane) {
    const u16* p = base + (size_t)(row + (lane & 15)) * ld + (k0 + ((lane >> 4) << 3));
    FragU u;
    u.q[0] = *(const uint4*)(p);
    u.q[1] = *(const uint4*)(p + 16);
    return u.v;
}

// B fragment (32x16 bf16) from a column-major-B == row-major-(N x K) "Bt" buffer.
// Lane layout: lanes 0-15 col N=lane, K {k0..k0+7, k0+16..k0+23}; lanes 16-31 K {+8.., +24..}.
__device__ __forceinline__ v16bf load_frag_b(const u16* base, int ld, int ncol, int k0, int lane) {
    const u16* p = base + (size_t)(ncol + (lane & 15)) * ld + (k0 + ((lane >> 4) << 3));
    FragU u;
    u.q[0] = *(const uint4*)(p);
    u.q[1] = *(const uint4*)(p + 16);
    return u.v;
}

__device__ __forceinline__ v8f wmma_bf16(v16bf a, v16bf b, v8f c) {
    return __builtin_amdgcn_wmma_f32_16x16x32_bf16(false, a, false, b, (short)0, c, false, false);
}

__device__ __forceinline__ float gelu_tanh(float v) {
    return 0.5f * v * (1.0f + tanhf(0.79788456080286535588f * (v + 0.044715f * v * v * v)));
}

// ---------------------------------------------------------------------------
// Weight transpose + f32->bf16:  Wt[n*K + k] = bf16(W[k*N + n])
// ---------------------------------------------------------------------------
__global__ void transpose_bf16_kernel(const float* __restrict__ W, u16* __restrict__ Wt,
                                      int K, int N) {
    size_t idx = (size_t)blockIdx.x * blockDim.x + threadIdx.x;
    size_t total = (size_t)K * N;
    if (idx >= total) return;
    int n = (int)(idx / K);
    int k = (int)(idx % K);
    Wt[idx] = f32_to_bf16(W[(size_t)k * N + n]);
}

__global__ void convert_bf16_kernel(const float* __restrict__ X, u16* __restrict__ Y, size_t n) {
    size_t idx = (size_t)blockIdx.x * blockDim.x + threadIdx.x;
    if (idx < n) Y[idx] = f32_to_bf16(X[idx]);
}

// ---------------------------------------------------------------------------
// LayerNorm over rows of length D (D multiple of 128), one wave per row,
// output bf16. Uses wave32 shfl_xor reductions.
// ---------------------------------------------------------------------------
__global__ void ln_rows_kernel(const float* __restrict__ X, const float* __restrict__ g,
                               const float* __restrict__ b, u16* __restrict__ Y,
                               int D, int nrows) {
    int wave = threadIdx.x >> 5;
    int lane = threadIdx.x & 31;
    int row  = blockIdx.x * 4 + wave;
    if (row >= nrows) return;
    const float4* xp = (const float4*)(X + (size_t)row * D);
    int nv = D >> 7;   // float4 chunks per lane (<=8 for D<=1024)
    float4 v[8];
    float s = 0.f, ss = 0.f;
    for (int i = 0; i < nv; ++i) {
        v[i] = xp[(size_t)i * 32 + lane];
        s  += v[i].x + v[i].y + v[i].z + v[i].w;
        ss += v[i].x*v[i].x + v[i].y*v[i].y + v[i].z*v[i].z + v[i].w*v[i].w;
    }
    for (int m = 16; m >= 1; m >>= 1) {
        s  += __shfl_xor(s,  m, 32);
        ss += __shfl_xor(ss, m, 32);
    }
    float mu  = s / (float)D;
    float var = ss / (float)D - mu * mu;
    float r   = rsqrtf(var + 1e-5f);
    u16* yp = Y + (size_t)row * D;
    for (int i = 0; i < nv; ++i) {
        int idx = (i * 32 + lane) * 4;
        float xv[4] = {v[i].x, v[i].y, v[i].z, v[i].w};
        #pragma unroll
        for (int c = 0; c < 4; ++c)
            yp[idx + c] = f32_to_bf16((xv[c] - mu) * r * g[idx + c] + b[idx + c]);
    }
}

// ---------------------------------------------------------------------------
// Generic WMMA GEMM: C(MxN) = A(MxK, bf16 row-major) * Bt(NxK, bf16)^T
// Block = 128 threads (4 waves). Wave computes 32x64 (8 accumulators, ~130
// VGPRs total -> no spills, K-loop unrolls & pipelines). Block tile: 32x256.
// Epilogue selected by EPI:
//   0: Cf = acc
//   2: Cf = acc, Cb = bf16(acc)
//   3: Cf = acc + addend
//   4: Cb = bf16(gelu(acc + bias[col]))
//   5: Cf = acc + bias[col] + addend
//   6: Cb stored transposed per-batch: Vt[(b*N + col)*256 + t], row = b*256+t
// ---------------------------------------------------------------------------
template <int EPI>
__global__ __launch_bounds__(128)
void gemm_wmma(const u16* __restrict__ A, const u16* __restrict__ Bt,
               float* __restrict__ Cf, u16* __restrict__ Cb,
               const float* __restrict__ bias, const float* __restrict__ addend,
               int M, int N, int K) {
    int lane = threadIdx.x & 31;
    int w    = threadIdx.x >> 5;
    int nbase = blockIdx.x * 256 + w * 64;
    int m0    = blockIdx.y * 32;

    AccU acc[2][4];
    #pragma unroll
    for (int i = 0; i < 2; ++i)
        #pragma unroll
        for (int j = 0; j < 4; ++j)
            #pragma unroll
            for (int e = 0; e < 8; ++e) acc[i][j].f[e] = 0.f;

    for (int k0 = 0; k0 < K; k0 += 32) {
        v16bf a0 = load_frag_a(A, K, m0,      k0, lane);
        v16bf a1 = load_frag_a(A, K, m0 + 16, k0, lane);
        #pragma unroll
        for (int nt = 0; nt < 4; ++nt) {
            v16bf bf = load_frag_b(Bt, K, nbase + nt * 16, k0, lane);
            acc[0][nt].v = wmma_bf16(a0, bf, acc[0][nt].v);
            acc[1][nt].v = wmma_bf16(a1, bf, acc[1][nt].v);
        }
    }

    // Epilogue. C layout: col = n0 + (lane&15); rows m0 + (lane<16?0:8) + j.
    #pragma unroll
    for (int mi = 0; mi < 2; ++mi) {
        #pragma unroll
        for (int nt = 0; nt < 4; ++nt) {
            int col   = nbase + nt * 16 + (lane & 15);
            int rbase = m0 + mi * 16 + ((lane >> 4) << 3);
            #pragma unroll
            for (int j = 0; j < 8; ++j) {
                int row = rbase + j;
                size_t idx = (size_t)row * N + col;
                float v = acc[mi][nt].f[j];
                if (EPI == 0) {
                    Cf[idx] = v;
                } else if (EPI == 2) {
                    Cf[idx] = v;
                    Cb[idx] = f32_to_bf16(v);
                } else if (EPI == 3) {
                    Cf[idx] = v + addend[idx];
                } else if (EPI == 4) {
                    Cb[idx] = f32_to_bf16(gelu_tanh(v + bias[col]));
                } else if (EPI == 5) {
                    Cf[idx] = v + bias[col] + addend[idx];
                } else if (EPI == 6) {
                    int bb = row >> 8;          // batch  (T = 256 rows per batch)
                    int tt = row & 255;         // token
                    Cb[((size_t)bb * N + col) * 256 + tt] = f32_to_bf16(v);
                }
            }
        }
    }
}

// ---------------------------------------------------------------------------
// Fused attention: per (b, h), scores = Qh Kh^T * 1/8, softmax over 256,
// attn = P Vh.  Block = 128 threads (4 waves); each wave owns 16 query rows.
// Qb: (B,L,1024) bf16; Kb: (B,T,1024) bf16 (post-LN); Vt: (B,1024,T) bf16;
// attnb out: (B,L,1024) bf16.
// ---------------------------------------------------------------------------
__global__ __launch_bounds__(128)
void attention_kernel(const u16* __restrict__ Qb, const u16* __restrict__ Kb,
                      const u16* __restrict__ Vt, u16* __restrict__ attnb) {
    constexpr int Lq = 1024, D = 1024, T = 256, HEAD = 64;
    __shared__ __align__(16) u16 Plds[4][16][256];

    int lane = threadIdx.x & 31;
    int w    = threadIdx.x >> 5;
    int h    = blockIdx.y;
    int b    = blockIdx.z;
    int l0   = blockIdx.x * 64 + w * 16;

    const u16* Qbase = Qb + (size_t)b * Lq * D;
    const u16* Kbase = Kb + (size_t)b * T * D;
    const u16* Vbase = Vt + (size_t)b * D * T;

    // ---- scores: 16 x 256 per wave (16 tiles), K = HEAD = 64 (2 steps) ----
    AccU S[16];
    #pragma unroll
    for (int nt = 0; nt < 16; ++nt)
        #pragma unroll
        for (int e = 0; e < 8; ++e) S[nt].f[e] = 0.f;

    v16bf aq0 = load_frag_a(Qbase, D, l0, h * HEAD,      lane);
    v16bf aq1 = load_frag_a(Qbase, D, l0, h * HEAD + 32, lane);
    #pragma unroll
    for (int nt = 0; nt < 16; ++nt) {
        v16bf bk0 = load_frag_b(Kbase, D, nt * 16, h * HEAD,      lane);
        v16bf bk1 = load_frag_b(Kbase, D, nt * 16, h * HEAD + 32, lane);
        S[nt].v = wmma_bf16(aq0, bk0, S[nt].v);
        S[nt].v = wmma_bf16(aq1, bk1, S[nt].v);
    }

    const float scale = 0.125f;   // 1/sqrt(64)
    #pragma unroll
    for (int nt = 0; nt < 16; ++nt)
        #pragma unroll
        for (int e = 0; e < 8; ++e) S[nt].f[e] *= scale;

    // ---- softmax over 256 columns: local over tiles, then 16-lane halves ----
    float rmax[8], rsum[8];
    #pragma unroll
    for (int j = 0; j < 8; ++j) {
        float m = -3.0e38f;
        #pragma unroll
        for (int nt = 0; nt < 16; ++nt) m = fmaxf(m, S[nt].f[j]);
        #pragma unroll
        for (int msk = 8; msk >= 1; msk >>= 1) m = fmaxf(m, __shfl_xor(m, msk, 32));
        rmax[j] = m;
    }
    #pragma unroll
    for (int j = 0; j < 8; ++j) {
        float s = 0.f;
        #pragma unroll
        for (int nt = 0; nt < 16; ++nt) {
            float e = __expf(S[nt].f[j] - rmax[j]);
            S[nt].f[j] = e;
            s += e;
        }
        #pragma unroll
        for (int msk = 8; msk >= 1; msk >>= 1) s += __shfl_xor(s, msk, 32);
        rsum[j] = 1.0f / s;
    }

    // stage probabilities to LDS as bf16 (row-major 16 x 256 per wave)
    #pragma unroll
    for (int nt = 0; nt < 16; ++nt) {
        int col = nt * 16 + (lane & 15);
        int rb  = (lane >> 4) << 3;
        #pragma unroll
        for (int j = 0; j < 8; ++j)
            Plds[w][rb + j][col] = f32_to_bf16(S[nt].f[j] * rsum[j]);
    }
    __syncthreads();

    // ---- attn = P(16x256) @ Vh(256x64): 4 output tiles, K loop of 8 ----
    AccU O[4];
    #pragma unroll
    for (int nt = 0; nt < 4; ++nt)
        #pragma unroll
        for (int e = 0; e < 8; ++e) O[nt].f[e] = 0.f;

    const u16* Pbase = &Plds[w][0][0];
    for (int kt = 0; kt < 8; ++kt) {
        v16bf ap = load_frag_a(Pbase, 256, 0, kt * 32, lane);
        #pragma unroll
        for (int nt = 0; nt < 4; ++nt) {
            v16bf bv = load_frag_b(Vbase, T, h * HEAD + nt * 16, kt * 32, lane);
            O[nt].v = wmma_bf16(ap, bv, O[nt].v);
        }
    }

    #pragma unroll
    for (int nt = 0; nt < 4; ++nt) {
        int col = h * HEAD + nt * 16 + (lane & 15);
        int rb  = l0 + ((lane >> 4) << 3);
        #pragma unroll
        for (int j = 0; j < 8; ++j)
            attnb[((size_t)b * Lq + rb + j) * D + col] = f32_to_bf16(O[nt].f[j]);
    }
}

// ---------------------------------------------------------------------------
// Host-side pipeline
// ---------------------------------------------------------------------------
extern "C" void kernel_launch(void* const* d_in, const int* in_sizes, int n_in,
                              void* d_out, int out_size, void* d_ws, size_t ws_size,
                              hipStream_t stream) {
    (void)in_sizes; (void)n_in; (void)out_size; (void)ws_size;
    constexpr int B = 16, T = 256, L = 1024, D = 1024, POS = 512, F = 4096;

    const float* grid_f  = (const float*)d_in[0];
    const float* qpos    = (const float*)d_in[1];
    const float* Wq      = (const float*)d_in[2];
    const float* Wk      = (const float*)d_in[3];
    const float* Wv      = (const float*)d_in[4];
    const float* Wo      = (const float*)d_in[5];
    const float* ln_g_g  = (const float*)d_in[6];
    const float* ln_g_b  = (const float*)d_in[7];
    const float* ln_q_g  = (const float*)d_in[8];
    const float* ln_q_b  = (const float*)d_in[9];
    const float* ln_m_g  = (const float*)d_in[10];
    const float* ln_m_b  = (const float*)d_in[11];
    const float* W1      = (const float*)d_in[12];
    const float* b1      = (const float*)d_in[13];
    const float* W2      = (const float*)d_in[14];
    const float* b2      = (const float*)d_in[15];
    float* out = (float*)d_out;

    // workspace carve-up (bytes, 256-aligned)
    char* w = (char*)d_ws;
    size_t cur = 0;
    auto alloc = [&](size_t bytes) {
        void* p = w + cur;
        cur += (bytes + 255) & ~(size_t)255;
        return p;
    };
    u16*   WqT   = (u16*)  alloc((size_t)D * POS * 2);       // (1024 x 512)
    u16*   WkT   = (u16*)  alloc((size_t)D * D * 2);
    u16*   WvT   = (u16*)  alloc((size_t)D * D * 2);
    u16*   WoT   = (u16*)  alloc((size_t)D * D * 2);
    u16*   W1T   = (u16*)  alloc((size_t)F * D * 2);         // (4096 x 1024)
    u16*   W2T   = (u16*)  alloc((size_t)D * F * 2);         // (1024 x 4096)
    u16*   gridb = (u16*)  alloc((size_t)B * T * D * 2);
    u16*   qnb   = (u16*)  alloc((size_t)B * L * POS * 2);
    float* Kpre  = (float*)alloc((size_t)B * T * D * 4);
    u16*   Kb    = (u16*)  alloc((size_t)B * T * D * 2);
    u16*   Vt    = (u16*)  alloc((size_t)B * D * T * 2);
    float* Qf    = (float*)alloc((size_t)B * L * D * 4);
    u16*   Qb    = (u16*)  alloc((size_t)B * L * D * 2);
    u16*   attnb = (u16*)  alloc((size_t)B * L * D * 2);
    float* xf    = (float*)alloc((size_t)B * L * D * 4);
    u16*   xnb   = (u16*)  alloc((size_t)B * L * D * 2);
    u16*   hb    = (u16*)  alloc((size_t)B * L * F * 2);

    auto cdiv = [](size_t a, size_t b) { return (unsigned)((a + b - 1) / b); };

    // 1) weight transpose/convert
    transpose_bf16_kernel<<<cdiv((size_t)POS * D, 256), 256, 0, stream>>>(Wq, WqT, POS, D);
    transpose_bf16_kernel<<<cdiv((size_t)D * D, 256),   256, 0, stream>>>(Wk, WkT, D, D);
    transpose_bf16_kernel<<<cdiv((size_t)D * D, 256),   256, 0, stream>>>(Wv, WvT, D, D);
    transpose_bf16_kernel<<<cdiv((size_t)D * D, 256),   256, 0, stream>>>(Wo, WoT, D, D);
    transpose_bf16_kernel<<<cdiv((size_t)D * F, 256),   256, 0, stream>>>(W1, W1T, D, F);
    transpose_bf16_kernel<<<cdiv((size_t)F * D, 256),   256, 0, stream>>>(W2, W2T, F, D);

    // 2) activations -> bf16 / LN(query)
    convert_bf16_kernel<<<cdiv((size_t)B * T * D, 256), 256, 0, stream>>>(grid_f, gridb, (size_t)B * T * D);
    ln_rows_kernel<<<(B * L) / 4, 128, 0, stream>>>(qpos, ln_q_g, ln_q_b, qnb, POS, B * L);

    // 3) projections
    gemm_wmma<0><<<dim3(D / 256, (B * T) / 32), 128, 0, stream>>>(
        gridb, WkT, Kpre, nullptr, nullptr, nullptr, B * T, D, D);
    ln_rows_kernel<<<(B * T) / 4, 128, 0, stream>>>(Kpre, ln_g_g, ln_g_b, Kb, D, B * T);
    gemm_wmma<6><<<dim3(D / 256, (B * T) / 32), 128, 0, stream>>>(
        gridb, WvT, nullptr, Vt, nullptr, nullptr, B * T, D, D);
    gemm_wmma<2><<<dim3(D / 256, (B * L) / 32), 128, 0, stream>>>(
        qnb, WqT, Qf, Qb, nullptr, nullptr, B * L, D, POS);

    // 4) attention (16 row-blocks of 64, H=16 heads, B=16)
    attention_kernel<<<dim3(L / 64, 16, B), 128, 0, stream>>>(Qb, Kb, Vt, attnb);

    // 5) output projection + residual: x = Q + attn @ Wo
    gemm_wmma<3><<<dim3(D / 256, (B * L) / 32), 128, 0, stream>>>(
        attnb, WoT, xf, nullptr, nullptr, Qf, B * L, D, D);

    // 6) MLP
    ln_rows_kernel<<<(B * L) / 4, 128, 0, stream>>>(xf, ln_m_g, ln_m_b, xnb, D, B * L);
    gemm_wmma<4><<<dim3(F / 256, (B * L) / 32), 128, 0, stream>>>(
        xnb, W1T, nullptr, hb, b1, nullptr, B * L, F, D);
    gemm_wmma<5><<<dim3(D / 256, (B * L) / 32), 128, 0, stream>>>(
        hb, W2T, out, nullptr, b2, xf, B * L, D, F);
}
